// BERTLayer_6562710028902
// MI455X (gfx1250) — 8-bit matrix operand order repaired and checked by reading, compiled, not run on hardware
//
#include <hip/hip_runtime.h>
#include <hip/hip_bf16.h>
#include <cstdint>

#define BB 8
#define TT 512
#define CC 1024
#define II 4096
#define HH 16
#define DH 64
#define BT (BB * TT)
#define C3 (3 * CC)

#define BPITCH 80

typedef __attribute__((ext_vector_type(8))) int v8i;

__device__ __forceinline__ float wave_max(float v) {
#pragma unroll
  for (int o = 16; o > 0; o >>= 1) v = fmaxf(v, __shfl_xor(v, o, 32));
  return v;
}
__device__ __forceinline__ float wave_sum(float v) {
#pragma unroll
  for (int o = 16; o > 0; o >>= 1) v += __shfl_xor(v, o, 32);
  return v;
}
__device__ __forceinline__ float scale_of(const unsigned* slot) {
  return fmaxf(__uint_as_float(*slot) * (1.0f / 127.0f), 1e-8f);
}

__global__ void init_slots_kernel(unsigned* slots) {
  if (threadIdx.x < 16) slots[threadIdx.x] = 0u;
}

__global__ void layernorm_absmax_kernel(const float* __restrict__ x,
                                        const float* __restrict__ g,
                                        const float* __restrict__ b,
                                        float* __restrict__ y,
                                        unsigned* __restrict__ amax) {
  const int row = blockIdx.x;
  const int tid = threadIdx.x;
  const int lane = tid & 31, wave = tid >> 5;
  const float* xr = x + (size_t)row * CC;
  __shared__ float rs[8], rq[8], mv[2], rm[8];

  float v[4];
  float s = 0.f, q = 0.f;
#pragma unroll
  for (int k = 0; k < 4; ++k) {
    v[k] = xr[tid + k * 256];
    s += v[k];
    q += v[k] * v[k];
  }
  s = wave_sum(s);
  q = wave_sum(q);
  if (lane == 0) { rs[wave] = s; rq[wave] = q; }
  __syncthreads();
  if (wave == 0) {
    float ts = (lane < 8) ? rs[lane] : 0.f;
    float tq = (lane < 8) ? rq[lane] : 0.f;
    ts = wave_sum(ts);
    tq = wave_sum(tq);
    if (lane == 0) { mv[0] = ts * (1.0f / CC); mv[1] = tq * (1.0f / CC); }
  }
  __syncthreads();
  const float mean = mv[0];
  const float var  = mv[1] - mean * mean;
  const float rstd = rsqrtf(var + 1e-5f);

  float lmax = 0.f;
  float* yr = y + (size_t)row * CC;
#pragma unroll
  for (int k = 0; k < 4; ++k) {
    const int c = tid + k * 256;
    const float o = (v[k] - mean) * rstd * g[c] + b[c];
    yr[c] = o;
    lmax = fmaxf(lmax, fabsf(o));
  }
  lmax = wave_max(lmax);
  if (lane == 0) rm[wave] = lmax;
  __syncthreads();
  if (tid == 0) {
    float m = rm[0];
    for (int i = 1; i < 8; ++i) m = fmaxf(m, rm[i]);
    atomicMax(amax, __float_as_uint(m));
  }
}

__global__ void absmax_kernel(const float* __restrict__ x, size_t n,
                              unsigned* __restrict__ amax) {
  __shared__ float rm[8];
  const int lane = threadIdx.x & 31, wave = threadIdx.x >> 5;
  size_t i = (size_t)blockIdx.x * blockDim.x + threadIdx.x;
  const size_t stride = (size_t)gridDim.x * blockDim.x;
  float lmax = 0.f;
  for (; i < n; i += stride) lmax = fmaxf(lmax, fabsf(x[i]));
  lmax = wave_max(lmax);
  if (lane == 0) rm[wave] = lmax;
  __syncthreads();
  if (threadIdx.x == 0) {
    float m = rm[0];
    for (int k = 1; k < 8; ++k) m = fmaxf(m, rm[k]);
    atomicMax(amax, __float_as_uint(m));
  }
}

__global__ void quant_kernel(const float* __restrict__ x, int8_t* __restrict__ q,
                             size_t n, const unsigned* __restrict__ amax) {
  const float s = scale_of(amax);
  const float inv = 1.0f / s;
  size_t i = (size_t)blockIdx.x * blockDim.x + threadIdx.x;
  const size_t stride = (size_t)gridDim.x * blockDim.x;
  for (; i < n; i += stride) {
    float r = rintf(x[i] * inv);
    r = fminf(fmaxf(r, -127.f), 127.f);
    q[i] = (int8_t)r;
  }
}

__global__ void gemm_i8_wmma_kernel(const int8_t* __restrict__ Aq,
                                    const int8_t* __restrict__ Wq,
                                    const float* __restrict__ bias,
                                    const unsigned* __restrict__ sA,
                                    const unsigned* __restrict__ sW,
                                    const float* __restrict__ residual,
                                    float* __restrict__ out,
                                    unsigned* __restrict__ amax,
                                    int N, int K, int do_gelu) {
  __shared__ int8_t Bsh[2][64 * BPITCH];

  const int tid  = threadIdx.x;
  const int lane = tid & 31, wave = tid >> 5;
  const int row0 = blockIdx.y * 128 + wave * 16;
  const int col0 = blockIdx.x * 64;

  const int bn  = tid >> 2;
  const int bkc = (tid & 3) * 16;
  const int8_t* bsrc = Wq + (size_t)(col0 + bn) * K + bkc;
  const unsigned bdst0 = (unsigned)(size_t)&Bsh[0][bn * BPITCH + bkc];
  const unsigned bdst1 = (unsigned)(size_t)&Bsh[1][bn * BPITCH + bkc];

  const int8_t* arow = Aq + (size_t)(row0 + (lane & 15)) * K + ((lane >> 4) * 16);
  const int nB  = lane & 31;
  const int kbB = (lane >> 4) * 16;
  const int8_t* bread = &Bsh[0][(lane & 15) * BPITCH + kbB];
  const int stageOff = 64 * BPITCH;

  union AF { v8i v; unsigned long long d[4]; };
  union BF { v8i v; int4 q[2]; };

  auto issueB = [&](unsigned dst, int k0) {
    asm volatile("global_load_async_to_lds_b128 %0, %1, off"
                 :: "v"(dst),
                    "v"((unsigned long long)(size_t)(bsrc + k0))
                 : "memory");
  };
  auto loadA = [&](int k0, AF& af) {
    af.d[0] = *(const unsigned long long*)(arow + k0);
    af.d[1] = *(const unsigned long long*)(arow + k0 + 8);
    af.d[2] = *(const unsigned long long*)(arow + k0 + 32);
    af.d[3] = *(const unsigned long long*)(arow + k0 + 40);
  };

  const v8i zero = {0, 0, 0, 0, 0, 0, 0, 0};
  v8i acc[4];
#pragma unroll
  for (int t = 0; t < 4; ++t) acc[t] = zero;

  auto mma4 = [&](int stage, AF& af) {
#pragma unroll
    for (int t = 0; t < 4; ++t) {
      BF bf;
      const int8_t* bp = bread + stage * stageOff + t * (16 * BPITCH);
      bf.q[0] = *(const int4*)(bp);
      bf.q[1] = *(const int4*)(bp + 32);
      acc[t] = __builtin_amdgcn_wmma_i32_16x16x64_iu8(
          true, af.v, true, bf.v, acc[t], false, false);
    }
  };

  AF a0, a1;
  issueB(bdst0, 0);
  loadA(0, a0);

  const int nk = K / 64;
  int ks = 0;
  for (; ks + 2 < nk; ks += 2) {
    issueB(bdst1, (ks + 1) * 64);
    loadA((ks + 1) * 64, a1);
    asm volatile("s_wait_asynccnt 0x1" ::: "memory");
    __syncthreads();
    mma4(0, a0);
    __syncthreads();
    issueB(bdst0, (ks + 2) * 64);
    loadA((ks + 2) * 64, a0);
    asm volatile("s_wait_asynccnt 0x1" ::: "memory");
    __syncthreads();
    mma4(1, a1);
    __syncthreads();
  }
  issueB(bdst1, (ks + 1) * 64);
  loadA((ks + 1) * 64, a1);
  asm volatile("s_wait_asynccnt 0x1" ::: "memory");
  __syncthreads();
  mma4(0, a0);
  asm volatile("s_wait_asynccnt 0x0" ::: "memory");
  __syncthreads();
  mma4(1, a1);

  const float ss = scale_of(sA) * scale_of(sW);
  const int mbase = row0 + ((lane >> 4) << 3);
  const int nbase = col0 + (lane & 15);
  float lmax = 0.f;
#pragma unroll
  for (int t = 0; t < 4; ++t) {
    const int n = nbase + t * 16;
    const float bn2 = bias[n];
#pragma unroll
    for (int r = 0; r < 8; ++r) {
      const int m = mbase + r;
      float y = (float)acc[t][r] * ss + bn2;
      if (do_gelu) y = 0.5f * y * (1.0f + erff(y * 0.70710678118654752f));
      if (residual) y += residual[(size_t)m * N + n];
      out[(size_t)m * N + n] = y;
      lmax = fmaxf(lmax, fabsf(y));
    }
  }
  if (amax) {
    lmax = wave_max(lmax);
    if (lane == 0) atomicMax(amax, __float_as_uint(lmax));
  }
  (void)nB;
}

__global__ void attention_kernel(const float* __restrict__ qkv,
                                 const float* __restrict__ mask,
                                 float* __restrict__ outp,
                                 unsigned* __restrict__ amax) {
  const int b = blockIdx.z, h = blockIdx.y;
  const int tid = threadIdx.x, lane = tid & 31, wave = tid >> 5;
  __shared__ float p[4][TT];
  __shared__ float qld[4][DH];
  float lmaxo = 0.f;

  for (int r = 0; r < 8; ++r) {
    const int qrow = blockIdx.x * 32 + wave * 8 + r;
    const float* qptr = qkv + (size_t)(b * TT + qrow) * C3 + h * DH;
    qld[wave][lane] = qptr[lane];
    qld[wave][lane + 32] = qptr[lane + 32];
    __syncthreads();

    float lmax = -3.0e38f;
    for (int jj = 0; jj < 16; ++jj) {
      const int j = jj * 32 + lane;
      const float* kptr = qkv + (size_t)(b * TT + j) * C3 + CC + h * DH;
      float s = 0.f;
#pragma unroll
      for (int d = 0; d < DH; ++d) s += qld[wave][d] * kptr[d];
      s = s * 0.125f + (1.0f - mask[b * TT + j]) * -10000.0f;
      p[wave][j] = s;
      lmax = fmaxf(lmax, s);
    }
    lmax = wave_max(lmax);

    float lsum = 0.f;
    for (int jj = 0; jj < 16; ++jj) {
      const int j = jj * 32 + lane;
      const float e = expf(p[wave][j] - lmax);
      p[wave][j] = e;
      lsum += e;
    }
    lsum = wave_sum(lsum);
    const float inv = 1.0f / lsum;
    __syncthreads();

    const int d0 = lane * 2;
    float a0 = 0.f, a1 = 0.f;
    for (int j = 0; j < TT; ++j) {
      const float pj = p[wave][j];
      const float* vptr = qkv + (size_t)(b * TT + j) * C3 + 2 * CC + h * DH;
      a0 += pj * vptr[d0];
      a1 += pj * vptr[d0 + 1];
    }
    a0 *= inv;
    a1 *= inv;
    float* orow = outp + (size_t)(b * TT + qrow) * CC + h * DH;
    orow[d0] = a0;
    orow[d0 + 1] = a1;
    lmaxo = fmaxf(lmaxo, fmaxf(fabsf(a0), fabsf(a1)));
    __syncthreads();
  }
  lmaxo = wave_max(lmaxo);
  if (lane == 0) atomicMax(amax, __float_as_uint(lmaxo));
}

extern "C" void kernel_launch(void* const* d_in, const int* in_sizes, int n_in,
                              void* d_out, int out_size, void* d_ws, size_t ws_size,
                              hipStream_t stream) {
  (void)in_sizes; (void)n_in; (void)out_size; (void)ws_size;

  const float* hs    = (const float*)d_in[0];
  const float* mask  = (const float*)d_in[1];
  const float* ln1_g = (const float*)d_in[2];
  const float* ln1_b = (const float*)d_in[3];
  const float* ln2_g = (const float*)d_in[4];
  const float* ln2_b = (const float*)d_in[5];
  const float* w_qkv = (const float*)d_in[6];
  const float* b_qkv = (const float*)d_in[7];
  const float* w_out = (const float*)d_in[8];
  const float* b_out = (const float*)d_in[9];
  const float* w1    = (const float*)d_in[10];
  const float* b1    = (const float*)d_in[11];
  const float* w2    = (const float*)d_in[12];
  const float* b2    = (const float*)d_in[13];
  float* outp = (float*)d_out;

  unsigned char* base = (unsigned char*)d_ws;
  unsigned* slots = (unsigned*)base;
  float* xln  = (float*)(base + 256);
  float* big  = xln + (size_t)BT * CC;
  float* qkvb = big;
  float* aact = big + (size_t)BT * C3;
  float* gbuf = big;
  float* hbuf = big + (size_t)BT * II;
  int8_t* qact = (int8_t*)(hbuf + (size_t)BT * CC);
  int8_t* qwt  = qact + (size_t)BT * II;

  init_slots_kernel<<<1, 16, 0, stream>>>(slots);

  layernorm_absmax_kernel<<<BT, 256, 0, stream>>>(hs, ln1_g, ln1_b, xln, slots + 0);
  absmax_kernel<<<512, 256, 0, stream>>>(w_qkv, (size_t)C3 * CC, slots + 1);
  quant_kernel<<<2048, 256, 0, stream>>>(xln,   qact, (size_t)BT * CC, slots + 0);
  quant_kernel<<<2048, 256, 0, stream>>>(w_qkv, qwt,  (size_t)C3 * CC, slots + 1);
  gemm_i8_wmma_kernel<<<dim3(C3 / 64, BT / 128), 256, 0, stream>>>(
      qact, qwt, b_qkv, slots + 0, slots + 1, nullptr, qkvb, nullptr, C3, CC, 0);

  attention_kernel<<<dim3(TT / 32, HH, BB), 128, 0, stream>>>(qkvb, mask, aact, slots + 2);

  absmax_kernel<<<512, 256, 0, stream>>>(w_out, (size_t)CC * CC, slots + 3);
  quant_kernel<<<2048, 256, 0, stream>>>(aact,  qact, (size_t)BT * CC, slots + 2);
  quant_kernel<<<2048, 256, 0, stream>>>(w_out, qwt,  (size_t)CC * CC, slots + 3);
  gemm_i8_wmma_kernel<<<dim3(CC / 64, BT / 128), 256, 0, stream>>>(
      qact, qwt, b_out, slots + 2, slots + 3, hs, hbuf, nullptr, CC, CC, 0);

  layernorm_absmax_kernel<<<BT, 256, 0, stream>>>(hbuf, ln2_g, ln2_b, xln, slots + 4);
  absmax_kernel<<<512, 256, 0, stream>>>(w1, (size_t)II * CC, slots + 5);
  quant_kernel<<<2048, 256, 0, stream>>>(xln, qact, (size_t)BT * CC, slots + 4);
  quant_kernel<<<2048, 256, 0, stream>>>(w1,  qwt,  (size_t)II * CC, slots + 5);
  gemm_i8_wmma_kernel<<<dim3(II / 64, BT / 128), 256, 0, stream>>>(
      qact, qwt, b1, slots + 4, slots + 5, nullptr, gbuf, slots + 6, II, CC, 1);

  absmax_kernel<<<512, 256, 0, stream>>>(w2, (size_t)CC * II, slots + 7);
  quant_kernel<<<4096, 256, 0, stream>>>(gbuf, qact, (size_t)BT * II, slots + 6);
  quant_kernel<<<2048, 256, 0, stream>>>(w2,   qwt,  (size_t)CC * II, slots + 7);
  gemm_i8_wmma_kernel<<<dim3(CC / 64, BT / 128), 256, 0, stream>>>(
      qact, qwt, b2, slots + 6, slots + 7, hbuf, outp, nullptr, CC, II, 0);
}
